// constrained_network_57329223467241
// MI455X (gfx1250) — matrix operand order repaired and checked by reading, not compiled, not run on hardware
//
#include <hip/hip_runtime.h>
#include <hip/hip_bf16.h>

typedef _Float16 half_t;
typedef __attribute__((ext_vector_type(16))) _Float16 v16h;
typedef __attribute__((ext_vector_type(8)))  _Float16 v8h;
typedef __attribute__((ext_vector_type(8)))  float    v8f;

#define PI_F 3.14159265358979323846f

__device__ inline v8f wmma16x16x32(v16h a, v16h b, v8f c) {
  // (neg_a, A, neg_b, B, c_mod, C, reuse_a, reuse_b)
  return __builtin_amdgcn_wmma_f32_16x16x32_f16(false, a, false, b, (short)0, c, false, false);
}

// A fragment: 16x32 f16 tile, row-major with row stride ld (multiple of 8), 16B-aligned base.
// Lanes 0-15: rows M=lane, K=0..7 & 16..23; lanes 16-31: K=8..15 & 24..31.
__device__ inline v16h ldA(const half_t* base, int ld) {
  int lane = threadIdx.x & 31;
  int row  = lane & 15;
  int kb   = (lane & 16) ? 8 : 0;
  const half_t* p = base + row * ld + kb;
  v8h lo = *(const v8h*)p;
  v8h hi = *(const v8h*)(p + 16);
  return __builtin_shufflevector(lo, hi, 0, 1, 2, 3, 4, 5, 6, 7,
                                 8, 9, 10, 11, 12, 13, 14, 15);
}

// B fragment from pre-swizzled weights: tile = kt*(N/16)+ct, each lane's 16 halves contiguous.
__device__ inline v16h ldBsw(const half_t* wbase, int tile) {
  int lane = threadIdx.x & 31;
  return *(const v16h*)(wbase + ((size_t)tile * 32 + lane) * 16);
}

__device__ inline void atomicAddF(float* p, float v) {
  unsafeAtomicAdd(p, v);   // hardware global_atomic_add_f32
}

__device__ inline float silu_f(float x) { return x / (1.f + __expf(-x)); }

// ---------------------------------------------------------------- weights -> swizzled f16
__device__ inline void swizzleB(const float* __restrict__ src, half_t* __restrict__ dst,
                                int N, int srcK, int idx) {
  int tile = idx >> 9, rem = idx & 511;
  int lane = rem >> 4, h = rem & 15;
  int ntc = N >> 4;
  int kt = tile / ntc, ct = tile % ntc;
  int col = ct * 16 + (lane & 15);
  int kg  = kt * 32 + ((lane & 16) ? 8 : 0) + (h < 8 ? h : h + 8);
  float v = (kg < srcK) ? src[kg * N + col] : 0.f;
  dst[idx] = (half_t)v;
}

__global__ void cvt_weights_kernel(const float* __restrict__ Wr0, const float* __restrict__ Wr1,
                                   const float* __restrict__ Wr2, const float* __restrict__ Wsc0,
                                   const float* __restrict__ Wsc1,
                                   half_t* r0, half_t* r1, half_t* r2, half_t* c0, half_t* c1) {
  int idx = blockIdx.x * blockDim.x + threadIdx.x;
  if (idx < 3 * 32 * 64) {   // W_r0: src (8,64), padded K->32
    int l = idx / 2048, r = idx % 2048;
    swizzleB(Wr0 + l * 8 * 64, r0 + l * 2048, 64, 8, r);
  }
  if (idx < 3 * 64 * 64) {   // W_r1: (64,64)
    int l = idx / 4096, r = idx % 4096;
    swizzleB(Wr1 + l * 4096, r1 + l * 4096, 64, 64, r);
  }
  if (idx < 3 * 64 * 128) {  // W_r2: (64,128)
    int l = idx / 8192, r = idx % 8192;
    swizzleB(Wr2 + l * 8192, r2 + l * 8192, 128, 64, r);
  }
  if (idx < 3 * 512 * 96) {  // W_sc0: (512,96)
    int l = idx / 49152, r = idx % 49152;
    swizzleB(Wsc0 + l * 49152, c0 + l * 49152, 96, 512, r);
  }
  if (idx < 3 * 256 * 32) {  // W_sc1: (256,32)
    int l = idx / 8192, r = idx % 8192;
    swizzleB(Wsc1 + l * 8192, c1 + l * 8192, 32, 256, r);
  }
}

// ---------------------------------------------------------------- Newton-Schulz (2x32)
__global__ void ns_kernel(const float* __restrict__ projM, float* __restrict__ Ms) {
  if (threadIdx.x != 0 || blockIdx.x != 0) return;
  float M0[32], M1[32];
  for (int c = 0; c < 32; ++c) { M0[c] = projM[c]; M1[c] = projM[32 + c]; }
  for (int it = 0; it < 10; ++it) {
    float g00 = 0.f, g01 = 0.f, g11 = 0.f;
    for (int c = 0; c < 32; ++c) { g00 += M0[c] * M0[c]; g01 += M0[c] * M1[c]; g11 += M1[c] * M1[c]; }
    float a00 = g00 - 1.f, a01 = g01, a11 = g11 - 1.f;
    for (int c = 0; c < 32; ++c) {
      float n0 = M0[c] - 0.5f * (a00 * M0[c] + a01 * M1[c]);
      float n1 = M1[c] - 0.5f * (a01 * M0[c] + a11 * M1[c]);
      M0[c] = n0; M1[c] = n1;
    }
  }
  for (int c = 0; c < 32; ++c) { Ms[c] = M0[c]; Ms[32 + c] = M1[c]; }
}

// ---------------------------------------------------------------- init: uplift + embed
__global__ void init_kernel(const float* __restrict__ x, const int* __restrict__ attr,
                            const float* __restrict__ embed, const float* __restrict__ Ms,
                            float* __restrict__ y, float* __restrict__ y_old,
                            float* __restrict__ x_cur, float* __restrict__ z, int n) {
  int nd = blockIdx.x * blockDim.x + threadIdx.x;
  if (nd >= n) return;
  int at = attr[nd];
  for (int a = 0; a < 8; ++a) z[nd * 8 + a] = embed[at * 8 + a];
  float xv[6];
  for (int j = 0; j < 6; ++j) { xv[j] = x[(size_t)nd * 6 + j]; x_cur[(size_t)nd * 6 + j] = xv[j]; }
  size_t yb = (size_t)nd * 160;
  for (int i = 0; i < 64; ++i) { y[yb + i] = 0.f; y_old[yb + i] = 0.f; }
  for (int i = 0; i < 32; ++i)
    for (int c = 0; c < 3; ++c) {
      float v = xv[c] * Ms[i] + xv[3 + c] * Ms[32 + i];
      y[yb + 64 + i * 3 + c] = v;
      y_old[yb + 64 + i * 3 + c] = v;
    }
}

// ---------------------------------------------------------------- per-layer node pre
#define NPB 32
template <bool FULL>
__global__ void node_pre_kernel(int node_base,
                                const float* __restrict__ y, const float* __restrict__ z,
                                const half_t* __restrict__ wsc0, const half_t* __restrict__ wsc1,
                                const float* __restrict__ Wl10, const float* __restrict__ Wl11,
                                float* __restrict__ sc0, float* __restrict__ sc1,
                                float* __restrict__ s1, float* __restrict__ v1,
                                float* __restrict__ a0, float* __restrict__ aA,
                                float* __restrict__ aC, int n) {
  __shared__ alignas(32) half_t At[96 * 256];   // 48 KB; sc0 phase uses 32x512 view
  int node0 = node_base + blockIdx.x * NPB;
  int tid = threadIdx.x, lane = tid & 31, wv = tid >> 5;

  // ---- build A = s (x) z   (32 x 512) f16, one v8h store per (row, i)
  for (int g = tid; g < NPB * 64; g += 256) {
    int row = g >> 6, i = g & 63;
    int nd = node0 + row;
    v8h out;
    if (FULL || nd < n) {
      float yi = y[(size_t)nd * 160 + i];
#pragma unroll
      for (int a = 0; a < 8; ++a) out[a] = (half_t)(yi * z[nd * 8 + a]);
    } else {
#pragma unroll
      for (int a = 0; a < 8; ++a) out[a] = (half_t)0.f;
    }
    *(v8h*)&At[row * 512 + i * 8] = out;
  }
  __syncthreads();

  // ---- sc0 = A @ Wsc0 (512 x 96), scale 1/sqrt(512); 12 tiles, 16 K-steps
  for (int t = wv; t < 12; t += 8) {
    int rt = t / 6, ct = t % 6;
    v8f acc = {};
#pragma unroll 4
    for (int ks = 0; ks < 16; ++ks) {
      v16h a = ldA(&At[(rt * 16) * 512 + ks * 32], 512);
      v16h b = ldBsw(wsc0, ks * 6 + ct);
      acc = wmma16x16x32(a, b, acc);
    }
    int m0 = (lane & 16) ? 8 : 0, nn = lane & 15;
#pragma unroll
    for (int r = 0; r < 8; ++r) {
      int nd = node0 + rt * 16 + m0 + r;
      if (FULL || nd < n) sc0[(size_t)nd * 96 + ct * 16 + nn] = acc[r] * 0.04419417382415922f;
    }
  }
  __syncthreads();

  // ---- sc1: all 3 channels fused as M=96 rows (row = c*32 + localnode), K=256
  for (int g = tid; g < 96 * 32; g += 256) {
    int row = g >> 5, i = g & 31;
    int c = row >> 5, ln = row & 31;
    int nd = node0 + ln;
    v8h out;
    if (FULL || nd < n) {
      float vi = y[(size_t)nd * 160 + 64 + i * 3 + c];
#pragma unroll
      for (int a = 0; a < 8; ++a) out[a] = (half_t)(vi * z[nd * 8 + a]);
    } else {
#pragma unroll
      for (int a = 0; a < 8; ++a) out[a] = (half_t)0.f;
    }
    *(v8h*)&At[row * 256 + i * 8] = out;
  }
  __syncthreads();
  for (int t = wv; t < 12; t += 8) {   // 6 row tiles x 2 col tiles
    int rt = t >> 1, ct = t & 1;
    v8f acc = {};
#pragma unroll
    for (int ks = 0; ks < 8; ++ks) {
      v16h a = ldA(&At[(rt * 16) * 256 + ks * 32], 256);
      v16h b = ldBsw(wsc1, ks * 2 + ct);
      acc = wmma16x16x32(a, b, acc);
    }
    int m0 = (lane & 16) ? 8 : 0, nn = lane & 15;
#pragma unroll
    for (int r = 0; r < 8; ++r) {
      int row = rt * 16 + m0 + r;
      int c = row >> 5, nd = node0 + (row & 31);
      if (FULL || nd < n) sc1[(size_t)nd * 96 + (ct * 16 + nn) * 3 + c] = acc[r] * 0.0625f;
    }
  }

  // ---- s1 = s @ Wl1_0 / 8
  for (int idx = tid; idx < NPB * 64; idx += 256) {
    int row = idx >> 6, o = idx & 63, nd = node0 + row;
    if (!FULL && nd >= n) continue;
    float acc = 0.f;
    for (int i = 0; i < 64; ++i) acc += y[(size_t)nd * 160 + i] * Wl10[i * 64 + o];
    s1[(size_t)nd * 64 + o] = acc * 0.125f;
  }
  // ---- v1 = v @ Wl1_1 / sqrt(32)
  for (int idx = tid; idx < NPB * 96; idx += 256) {
    int row = idx / 96, rem = idx % 96, o = rem / 3, c = rem % 3, nd = node0 + row;
    if (!FULL && nd >= n) continue;
    float acc = 0.f;
    for (int i = 0; i < 32; ++i) acc += y[(size_t)nd * 160 + 64 + i * 3 + c] * Wl11[i * 32 + o];
    v1[(size_t)nd * 96 + o * 3 + c] = acc * 0.17677669529663687f;
  }
  // ---- zero accumulators
  for (int idx = tid; idx < NPB * 320; idx += 256) {
    int row = idx / 320, j = idx % 320, nd = node0 + row;
    if (!FULL && nd >= n) continue;
    if (j < 32)       a0[(size_t)nd * 32 + j] = 0.f;
    else if (j < 224) aA[(size_t)nd * 192 + (j - 32)] = 0.f;
    else              aC[(size_t)nd * 96 + (j - 224)] = 0.f;
  }
}

// ---------------------------------------------------------------- fused edge kernel
template <bool FULL>
__global__ void edge_kernel(int e_base,
                            const float* __restrict__ x_cur, const int* __restrict__ esrc,
                            const int* __restrict__ edst,
                            const half_t* __restrict__ wr0, const half_t* __restrict__ wr1,
                            const half_t* __restrict__ wr2,
                            const float* __restrict__ s1, const float* __restrict__ v1,
                            float* __restrict__ a0, float* __restrict__ aA,
                            float* __restrict__ aC, int E) {
  __shared__ alignas(32) half_t sX[64 * 32];
  __shared__ alignas(32) half_t sH0[64 * 64];
  __shared__ alignas(32) half_t sH1[64 * 64];
  __shared__ alignas(16) float  sWout[64 * 128];
  __shared__ float  sEA[64 * 3];
  __shared__ int    sSrc[64], sDst[64];
  int e0 = e_base + blockIdx.x * 64;
  int tid = threadIdx.x, lane = tid & 31, wv = tid >> 5;

  // ---- geometry + radial basis (threads 0..63, one edge each)
  if (tid < 64) {
    int ge = e0 + tid;
    float ea0 = 0.f, ea1 = 0.f, ea2 = 0.f;
    int sidx = 0, didx = 0;
    float ef[8];
#pragma unroll
    for (int j = 0; j < 8; ++j) ef[j] = 0.f;
    if (FULL || ge < E) {
      sidx = esrc[ge]; didx = edst[ge];
      __builtin_prefetch(&s1[(size_t)sidx * 64], 0, 3);
      __builtin_prefetch(&v1[(size_t)sidx * 96], 0, 3);
      float ex = x_cur[(size_t)sidx * 6 + 3] - x_cur[(size_t)didx * 6 + 3];
      float ey = x_cur[(size_t)sidx * 6 + 4] - x_cur[(size_t)didx * 6 + 4];
      float ez = x_cur[(size_t)sidx * 6 + 5] - x_cur[(size_t)didx * 6 + 5];
      float el = sqrtf(ex * ex + ey * ey + ez * ez + 1e-12f);
      float inv_el = 1.f / el;
      const float MAXR = 3.5f;
      float pref = 2.1380899352993948f * inv_el;  // sqrt(2/MAXR)*sqrt(8)/el
#pragma unroll
      for (int j = 0; j < 8; ++j) ef[j] = pref * __sinf((float)(j + 1) * PI_F * el / MAXR);
      float u = 2.f * (el / MAXR - 1.f);
      float fc = (u > 0.f) ? 0.f : ((u < -1.f) ? 1.f : 0.5f * (1.f - __cosf(PI_F * u)));
      float shc = 1.7320508075688772f * inv_el * fc;
      ea0 = shc * ex; ea1 = shc * ey; ea2 = shc * ez;
    }
    sSrc[tid] = sidx; sDst[tid] = didx;
    sEA[tid * 3 + 0] = ea0; sEA[tid * 3 + 1] = ea1; sEA[tid * 3 + 2] = ea2;
    v8h e8;
#pragma unroll
    for (int j = 0; j < 8; ++j) e8[j] = (half_t)ef[j];
    v8h z8;
#pragma unroll
    for (int j = 0; j < 8; ++j) z8[j] = (half_t)0.f;
    *(v8h*)&sX[tid * 32 + 0]  = e8;
    *(v8h*)&sX[tid * 32 + 8]  = z8;
    *(v8h*)&sX[tid * 32 + 16] = z8;
    *(v8h*)&sX[tid * 32 + 24] = z8;
  }
  __syncthreads();

  // ---- GEMM1: (64x32)@(32x64), silu(d/sqrt(8))
  for (int t = wv; t < 16; t += 8) {
    int rt = t >> 2, ct = t & 3;
    v16h a = ldA(&sX[(rt * 16) * 32], 32);
    v16h b = ldBsw(wr0, ct);
    v8f acc = {};
    acc = wmma16x16x32(a, b, acc);
    int m0 = (lane & 16) ? 8 : 0, nn = lane & 15;
#pragma unroll
    for (int r = 0; r < 8; ++r) {
      float d = silu_f(acc[r] * 0.35355339059327373f);
      sH0[(rt * 16 + m0 + r) * 64 + ct * 16 + nn] = (half_t)d;
    }
  }
  __syncthreads();

  // ---- GEMM2: (64x64)@(64x64), silu(d/8)
  for (int t = wv; t < 16; t += 8) {
    int rt = t >> 2, ct = t & 3;
    v8f acc = {};
#pragma unroll
    for (int ks = 0; ks < 2; ++ks) {
      v16h a = ldA(&sH0[(rt * 16) * 64 + ks * 32], 64);
      v16h b = ldBsw(wr1, ks * 4 + ct);
      acc = wmma16x16x32(a, b, acc);
    }
    int m0 = (lane & 16) ? 8 : 0, nn = lane & 15;
#pragma unroll
    for (int r = 0; r < 8; ++r) {
      float d = silu_f(acc[r] * 0.125f);
      sH1[(rt * 16 + m0 + r) * 64 + ct * 16 + nn] = (half_t)d;
    }
  }
  __syncthreads();

  // ---- GEMM3: (64x64)@(64x128), w = d/8
  for (int t = wv; t < 32; t += 8) {
    int rt = t >> 3, ct = t & 7;
    v8f acc = {};
#pragma unroll
    for (int ks = 0; ks < 2; ++ks) {
      v16h a = ldA(&sH1[(rt * 16) * 64 + ks * 32], 64);
      v16h b = ldBsw(wr2, ks * 8 + ct);
      acc = wmma16x16x32(a, b, acc);
    }
    int m0 = (lane & 16) ? 8 : 0, nn = lane & 15;
#pragma unroll
    for (int r = 0; r < 8; ++r)
      sWout[(rt * 16 + m0 + r) * 128 + ct * 16 + nn] = acc[r] * 0.125f;
  }
  __syncthreads();

  // ---- messages + scatter (4 threads per edge)
  int le = tid >> 2, sub = tid & 3;
  int ge = e0 + le;
  if (FULL || ge < E) {
    int sidx = sSrc[le], didx = sDst[le];
    float ea0 = sEA[le * 3], ea1 = sEA[le * 3 + 1], ea2 = sEA[le * 3 + 2];

    // mA: channels sub*16 .. sub*16+15 (float4 loads)
    const float4* wv4p = (const float4*)&sWout[le * 128 + sub * 16];
    const float4* sv4p = (const float4*)(s1 + (size_t)sidx * 64 + sub * 16);
#pragma unroll
    for (int q = 0; q < 4; ++q) {
      float4 w4 = wv4p[q];
      float4 s4 = sv4p[q];
      float vals[4] = {w4.x * s4.x, w4.y * s4.y, w4.z * s4.z, w4.w * s4.w};
      int ib = sub * 16 + q * 4;
#pragma unroll
      for (int r = 0; r < 4; ++r) {
        atomicAddF(&aA[(size_t)didx * 192 + (ib + r) * 3 + 0], vals[r] * ea0);
        atomicAddF(&aA[(size_t)didx * 192 + (ib + r) * 3 + 1], vals[r] * ea1);
        atomicAddF(&aA[(size_t)didx * 192 + (ib + r) * 3 + 2], vals[r] * ea2);
      }
    }

    // mB, mC: channels sub*8 .. sub*8+7 ; vv rows loaded as 6 float4's
    float vv[24];
    const float4* vp = (const float4*)(v1 + (size_t)sidx * 96 + sub * 24);
#pragma unroll
    for (int q = 0; q < 6; ++q) {
      float4 t4 = vp[q];
      vv[q * 4 + 0] = t4.x; vv[q * 4 + 1] = t4.y; vv[q * 4 + 2] = t4.z; vv[q * 4 + 3] = t4.w;
    }
    const float4* wb4 = (const float4*)&sWout[le * 128 + 64 + sub * 8];
    const float4* wc4 = (const float4*)&sWout[le * 128 + 96 + sub * 8];
    float wB[8], wC[8];
    {
      float4 t0 = wb4[0], t1 = wb4[1];
      wB[0] = t0.x; wB[1] = t0.y; wB[2] = t0.z; wB[3] = t0.w;
      wB[4] = t1.x; wB[5] = t1.y; wB[6] = t1.z; wB[7] = t1.w;
      float4 u0 = wc4[0], u1 = wc4[1];
      wC[0] = u0.x; wC[1] = u0.y; wC[2] = u0.z; wC[3] = u0.w;
      wC[4] = u1.x; wC[5] = u1.y; wC[6] = u1.z; wC[7] = u1.w;
    }
#pragma unroll
    for (int r = 0; r < 8; ++r) {
      int i = sub * 8 + r;
      float v0 = vv[r * 3 + 0], vy = vv[r * 3 + 1], v2 = vv[r * 3 + 2];
      float dot = v0 * ea0 + vy * ea1 + v2 * ea2;
      atomicAddF(&a0[(size_t)didx * 32 + i], wB[r] * dot);
      atomicAddF(&aC[(size_t)didx * 96 + i * 3 + 0], wC[r] * (vy * ea2 - v2 * ea1));
      atomicAddF(&aC[(size_t)didx * 96 + i * 3 + 1], wC[r] * (v2 * ea0 - v0 * ea2));
      atomicAddF(&aC[(size_t)didx * 96 + i * 3 + 2], wC[r] * (v0 * ea1 - vy * ea0));
    }
  }
}

// ---------------------------------------------------------------- per-layer node post
template <bool FULL>
__global__ void node_post_kernel(int node_base,
                                 const float* __restrict__ sc0, const float* __restrict__ sc1,
                                 const float* __restrict__ a0, const float* __restrict__ aA,
                                 const float* __restrict__ aC,
                                 const float* __restrict__ Wl20, const float* __restrict__ Wl21,
                                 const float* __restrict__ Wsi0, const float* __restrict__ Wsi1,
                                 const float* __restrict__ Ms,
                                 float* __restrict__ y, float* __restrict__ y_old,
                                 float* __restrict__ x_cur, float* __restrict__ x_out, int n) {
  __shared__ float sO0[8][96];
  __shared__ float sScal[8][64];
  __shared__ float sGated[8][96];
  __shared__ float sYu[8][160];
  int tid = threadIdx.x, lane = tid & 31, wv = tid >> 5;
  int nd = node_base + blockIdx.x * 8 + wv;
  bool act = FULL || (nd < n);
  const float inv = 0.17677669529663687f;  // 1/sqrt(32) segment scale

  if (act) {
    for (int j = 0; j < 3; ++j) {
      int o = lane * 3 + j;
      float s = 0.f;
      for (int i = 0; i < 32; ++i) s += a0[(size_t)nd * 32 + i] * Wl20[i * 96 + o];
      sO0[wv][o] = sc0[(size_t)nd * 96 + o] + s * inv * 0.17677669529663687f;
    }
  }
  __syncthreads();
  if (act) {
    for (int j = 0; j < 2; ++j) { int o = lane * 2 + j; sScal[wv][o] = silu_f(sO0[wv][o]); }
    {
      int o = lane;
      float g = 1.f / (1.f + __expf(-sO0[wv][64 + o]));
      for (int c = 0; c < 3; ++c) {
        float s = 0.f;
        for (int i = 0; i < 64; ++i) s += aA[(size_t)nd * 192 + i * 3 + c] * Wl21[i * 32 + o];
        for (int i = 0; i < 32; ++i) s += aC[(size_t)nd * 96 + i * 3 + c] * Wl21[(64 + i) * 32 + o];
        float o1 = sc1[(size_t)nd * 96 + o * 3 + c] + s * inv * 0.10206207261596575f; // /sqrt(96)
        sGated[wv][o * 3 + c] = o1 * g;
      }
    }
  }
  __syncthreads();
  if (act) {
    size_t yb = (size_t)nd * 160;
    for (int j = 0; j < 2; ++j) {
      int o = lane * 2 + j;
      float s = 0.f;
      for (int i = 0; i < 64; ++i) s += sScal[wv][i] * Wsi0[i * 64 + o];
      float yv = y[yb + o];
      float up = 2.f * yv - y_old[yb + o] + 0.1f * (s * 0.125f);
      sYu[wv][o] = up; y[yb + o] = up; y_old[yb + o] = yv;
    }
    {
      int o = lane;
      for (int c = 0; c < 3; ++c) {
        float s = 0.f;
        for (int i = 0; i < 32; ++i) s += sGated[wv][i * 3 + c] * Wsi1[i * 32 + o];
        int j = 64 + o * 3 + c;
        float yv = y[yb + j];
        float up = 2.f * yv - y_old[yb + j] + 0.1f * (s * 0.17677669529663687f);
        sYu[wv][j] = up; y[yb + j] = up; y_old[yb + j] = yv;
      }
    }
  }
  __syncthreads();
  if (act && lane < 6) {
    int k = lane / 3, c = lane % 3;
    float s = 0.f;
    for (int i = 0; i < 32; ++i) s += sYu[wv][64 + i * 3 + c] * Ms[k * 32 + i];
    x_cur[(size_t)nd * 6 + lane] = s;
    if (x_out) x_out[(size_t)nd * 6 + lane] = s;
  }
}

// ---------------------------------------------------------------- host launcher
extern "C" void kernel_launch(void* const* d_in, const int* in_sizes, int n_in,
                              void* d_out, int out_size, void* d_ws, size_t ws_size,
                              hipStream_t stream) {
  const float* x        = (const float*)d_in[0];
  const int*   nodeattr = (const int*)d_in[1];
  const int*   esrc     = (const int*)d_in[2];
  const int*   edst     = (const int*)d_in[3];
  const float* projM    = (const float*)d_in[4];
  const float* embed    = (const float*)d_in[5];
  const float* W_sc0    = (const float*)d_in[6];
  const float* W_sc1    = (const float*)d_in[7];
  const float* W_l1_0   = (const float*)d_in[8];
  const float* W_l1_1   = (const float*)d_in[9];
  const float* W_r0     = (const float*)d_in[10];
  const float* W_r1     = (const float*)d_in[11];
  const float* W_r2     = (const float*)d_in[12];
  const float* W_l2_0   = (const float*)d_in[13];
  const float* W_l2_1   = (const float*)d_in[14];
  const float* W_si0    = (const float*)d_in[15];
  const float* W_si1    = (const float*)d_in[16];

  int n = in_sizes[0] / 6;
  int E = in_sizes[2];

  char* p = (char*)d_ws;
  auto carve = [&](size_t bytes) -> char* {
    char* r = p;
    p += (bytes + 255) & ~(size_t)255;
    return r;
  };
  float* Ms    = (float*)carve(64 * 4);
  float* y     = (float*)carve((size_t)n * 160 * 4);
  float* y_old = (float*)carve((size_t)n * 160 * 4);
  float* x_cur = (float*)carve((size_t)n * 6 * 4);
  float* z     = (float*)carve((size_t)n * 8 * 4);
  float* s1    = (float*)carve((size_t)n * 64 * 4);
  float* v1    = (float*)carve((size_t)n * 96 * 4);
  float* sc0   = (float*)carve((size_t)n * 96 * 4);
  float* sc1   = (float*)carve((size_t)n * 96 * 4);
  float* a0    = (float*)carve((size_t)n * 32 * 4);
  float* aA    = (float*)carve((size_t)n * 192 * 4);
  float* aC    = (float*)carve((size_t)n * 96 * 4);
  half_t* h_r0  = (half_t*)carve(3 * 32 * 64 * 2);
  half_t* h_r1  = (half_t*)carve(3 * 64 * 64 * 2);
  half_t* h_r2  = (half_t*)carve(3 * 64 * 128 * 2);
  half_t* h_sc0 = (half_t*)carve(3 * 512 * 96 * 2);
  half_t* h_sc1 = (half_t*)carve(3 * 256 * 32 * 2);

  cvt_weights_kernel<<<(3 * 512 * 96 + 255) / 256, 256, 0, stream>>>(
      W_r0, W_r1, W_r2, W_sc0, W_sc1, h_r0, h_r1, h_r2, h_sc0, h_sc1);
  ns_kernel<<<1, 32, 0, stream>>>(projM, Ms);
  init_kernel<<<(n + 255) / 256, 256, 0, stream>>>(x, nodeattr, embed, Ms, y, y_old, x_cur, z, n);

  int nfull_pre  = n / NPB;           // full 32-node blocks
  int ntail_pre  = n - nfull_pre * NPB;
  int efull      = E / 64;            // full 64-edge blocks
  int etail      = E - efull * 64;
  int nfull_post = n / 8;             // full 8-node blocks
  int ntail_post = n - nfull_post * 8;

  for (int l = 0; l < 3; ++l) {
    const half_t* wsc0l = h_sc0 + (size_t)l * 512 * 96;
    const half_t* wsc1l = h_sc1 + (size_t)l * 256 * 32;
    const float*  wl10l = W_l1_0 + (size_t)l * 64 * 64;
    const float*  wl11l = W_l1_1 + (size_t)l * 32 * 32;
    if (nfull_pre)
      node_pre_kernel<true><<<nfull_pre, 256, 0, stream>>>(
          0, y, z, wsc0l, wsc1l, wl10l, wl11l, sc0, sc1, s1, v1, a0, aA, aC, n);
    if (ntail_pre)
      node_pre_kernel<false><<<1, 256, 0, stream>>>(
          nfull_pre * NPB, y, z, wsc0l, wsc1l, wl10l, wl11l, sc0, sc1, s1, v1, a0, aA, aC, n);

    const half_t* wr0l = h_r0 + (size_t)l * 32 * 64;
    const half_t* wr1l = h_r1 + (size_t)l * 64 * 64;
    const half_t* wr2l = h_r2 + (size_t)l * 64 * 128;
    if (efull)
      edge_kernel<true><<<efull, 256, 0, stream>>>(
          0, x_cur, esrc, edst, wr0l, wr1l, wr2l, s1, v1, a0, aA, aC, E);
    if (etail)
      edge_kernel<false><<<1, 256, 0, stream>>>(
          efull * 64, x_cur, esrc, edst, wr0l, wr1l, wr2l, s1, v1, a0, aA, aC, E);

    const float* wl20l = W_l2_0 + (size_t)l * 32 * 96;
    const float* wl21l = W_l2_1 + (size_t)l * 96 * 32;
    const float* wsi0l = W_si0 + (size_t)l * 64 * 64;
    const float* wsi1l = W_si1 + (size_t)l * 32 * 32;
    float* xo = (l == 2) ? (float*)d_out : nullptr;
    if (nfull_post)
      node_post_kernel<true><<<nfull_post, 256, 0, stream>>>(
          0, sc0, sc1, a0, aA, aC, wl20l, wl21l, wsi0l, wsi1l, Ms, y, y_old, x_cur, xo, n);
    if (ntail_post)
      node_post_kernel<false><<<1, 256, 0, stream>>>(
          nfull_post * 8, sc0, sc1, a0, aA, aC, wl20l, wl21l, wsi0l, wsi1l, Ms, y, y_old, x_cur, xo, n);
  }
}
